// SimpleGNN_65300682768499
// MI455X (gfx1250) — matrix operand order repaired and checked by reading, not compiled, not run on hardware
//
#include <hip/hip_runtime.h>
#include <hip/hip_bf16.h>
#include <math.h>

// GCN (2 layers) + edge MLP for MI455X (gfx1250, wave32).
// Dense parts (x@W1, ef@We1) -> v_wmma_f32_16x16x32_f16.
// Irregular aggregation -> f32 global atomics (bandwidth bound, ~2GB total).

typedef __attribute__((ext_vector_type(16))) _Float16 v16h;
typedef __attribute__((ext_vector_type(8)))  float    v8f;

// ---------------------------------------------------------------- elementwise

__global__ void k_fill1(float* __restrict__ p, int n) {
  int i = blockIdx.x * blockDim.x + threadIdx.x;
  if (i < n) p[i] = 1.0f;
}

__global__ void k_deg_accum(const int* __restrict__ col, float* __restrict__ deg, int E) {
  int e = blockIdx.x * blockDim.x + threadIdx.x;
  if (e < E) atomicAdd(&deg[col[e]], 1.0f);
}

__global__ void k_rsqrt(float* __restrict__ d, int n) {
  int i = blockIdx.x * blockDim.x + threadIdx.x;
  if (i < n) d[i] = rsqrtf(d[i]);  // deg >= 1 always (self loops)
}

// acc[n,f] = dinv[n]^2 * hlin[n,f]   (self-loop contribution, also zero-inits acc)
__global__ void k_selfloop16(const float* __restrict__ dinv, const float* __restrict__ hlin,
                             float* __restrict__ acc, int total) {
  int i = blockIdx.x * blockDim.x + threadIdx.x;
  if (i < total) {
    int n = i >> 4;
    float dv = dinv[n];
    acc[i] = dv * dv * hlin[i];
  }
}

// acc[col,f] += dinv[row]*dinv[col]*hlin[row,f]   (thread per (edge,f), f = tid&15)
__global__ void k_edge_agg16(const int* __restrict__ row, const int* __restrict__ col,
                             const float* __restrict__ dinv, const float* __restrict__ hlin,
                             float* __restrict__ acc, long total) {
  long tid = (long)blockIdx.x * blockDim.x + threadIdx.x;
  if (tid < total) {
    int e = (int)(tid >> 4), f = (int)(tid & 15);
    int s = row[e], d = col[e];
    float nrm = dinv[s] * dinv[d];
    atomicAdd(&acc[(long)d * 16 + f], nrm * hlin[(long)s * 16 + f]);
  }
}

__global__ void k_relu_bias16(float* __restrict__ h, const float* __restrict__ b1, int total) {
  int i = blockIdx.x * blockDim.x + threadIdx.x;
  if (i < total) h[i] = fmaxf(h[i] + b1[i & 15], 0.0f);
}

// hw2[n,c] = sum_f h[n,f] * W2[f,c]   (16x2 GEMM, trivial)
__global__ void k_hw2(const float* __restrict__ h, const float* __restrict__ W2,
                      float* __restrict__ hw2, int total) {
  int tid = blockIdx.x * blockDim.x + threadIdx.x;
  if (tid < total) {
    int n = tid >> 1, c = tid & 1;
    const float* hr = h + (long)n * 16;
    float s = 0.0f;
#pragma unroll
    for (int f = 0; f < 16; ++f) s += hr[f] * W2[f * 2 + c];
    hw2[tid] = s;
  }
}

__global__ void k_selfloop2(const float* __restrict__ dinv, const float* __restrict__ hw2,
                            float* __restrict__ out, int total) {
  int i = blockIdx.x * blockDim.x + threadIdx.x;
  if (i < total) {
    int n = i >> 1;
    float dv = dinv[n];
    out[i] = dv * dv * hw2[i];
  }
}

__global__ void k_edge_agg2(const int* __restrict__ row, const int* __restrict__ col,
                            const float* __restrict__ dinv, const float* __restrict__ hw2,
                            float* __restrict__ out, long total) {
  long tid = (long)blockIdx.x * blockDim.x + threadIdx.x;
  if (tid < total) {
    int e = (int)(tid >> 1), c = (int)(tid & 1);
    int s = row[e], d = col[e];
    float nrm = dinv[s] * dinv[d];
    atomicAdd(&out[(long)d * 2 + c], nrm * hw2[(long)s * 2 + c]);
  }
}

__global__ void k_bias2(float* __restrict__ out, const float* __restrict__ b2, int total) {
  int i = blockIdx.x * blockDim.x + threadIdx.x;
  if (i < total) out[i] += b2[i & 1];
}

// ---------------------------------------------------------------- WMMA GEMM 1
// hlin[100000,16] = x[100000,128] @ W1[128,16], f16 inputs, f32 accumulate.
// One wave per 16-row tile; K=128 as 4 chunks of 32; B (=W1) held in regs.

__device__ __forceinline__ void pack8(v16h& A, int base, float4 a, float4 b) {
  A[base + 0] = (_Float16)a.x; A[base + 1] = (_Float16)a.y;
  A[base + 2] = (_Float16)a.z; A[base + 3] = (_Float16)a.w;
  A[base + 4] = (_Float16)b.x; A[base + 5] = (_Float16)b.y;
  A[base + 6] = (_Float16)b.z; A[base + 7] = (_Float16)b.w;
}

__global__ void k_gemm1_wmma(const float* __restrict__ x, const float* __restrict__ W1,
                             float* __restrict__ hlin, int N, int ntiles) {
  const int lane = threadIdx.x & 31;
  const int wid  = (blockIdx.x * blockDim.x + threadIdx.x) >> 5;
  const int nw   = (gridDim.x * blockDim.x) >> 5;
  const int n    = lane & 15;
  const int half = lane >> 4;

  // B layout: lane = N column; elems 0-7 K=half*8+j, elems 8-15 K=16+half*8+j.
  v16h B[4];
#pragma unroll
  for (int c = 0; c < 4; ++c)
#pragma unroll
    for (int e = 0; e < 16; ++e)
      B[c][e] = (_Float16)W1[(c * 32 + half * 8 + e + (e & 8)) * 16 + n];

  for (int t = wid; t < ntiles; t += nw) {
    const int m0 = t * 16;
    int m = m0 + n; if (m >= N) m = N - 1;        // load clamp (tail only)
    const float* xr = x + (long)m * 128;
    v8f acc = {};
#pragma unroll
    for (int c = 0; c < 4; ++c) {
      // A layout: lane = M row; elems 0-7 K=half*8+j, elems 8-15 K=16+half*8+j.
      const float4* p0 = (const float4*)(xr + c * 32 + half * 8);
      const float4* p1 = (const float4*)(xr + c * 32 + 16 + half * 8);
      v16h A;
      pack8(A, 0, p0[0], p0[1]);
      pack8(A, 8, p1[0], p1[1]);
      acc = __builtin_amdgcn_wmma_f32_16x16x32_f16(false, A, false, B[c],
                                                   (short)0, acc, false, false);
    }
    // D layout: lane = N column; element j -> row half*8+j.
    float* dst = hlin + (long)(m0 + half * 8) * 16 + n;
    if (m0 + 16 <= N) {                            // wave-uniform fast path
#pragma unroll
      for (int j = 0; j < 8; ++j) dst[j * 16] = acc[j];
    } else {
#pragma unroll
      for (int j = 0; j < 8; ++j)
        if (m0 + half * 8 + j < N) dst[j * 16] = acc[j];
    }
  }
}

// ------------------------------------------------------------ WMMA edge MLP
// Per 16-edge tile: A = [h[s]|h[d]] 16x32 f16 gathered, B = We1 32x16 in regs,
// then relu(+be1), dot with We2 via xor-shuffle reduction, sigmoid(+be2).

__global__ void k_edge_mlp_wmma(const float* __restrict__ h, const int* __restrict__ row,
                                const int* __restrict__ col, const float* __restrict__ We1,
                                const float* __restrict__ be1, const float* __restrict__ We2,
                                const float* __restrict__ be2p, float* __restrict__ eout,
                                int E, int ntiles) {
  const int lane = threadIdx.x & 31;
  const int wid  = (blockIdx.x * blockDim.x + threadIdx.x) >> 5;
  const int nw   = (gridDim.x * blockDim.x) >> 5;
  const int n    = lane & 15;
  const int half = lane >> 4;

  v16h B;
#pragma unroll
  for (int e = 0; e < 16; ++e)
    B[e] = (_Float16)We1[(half * 8 + e + (e & 8)) * 16 + n];
  const float w2 = We2[n];
  const float bb = be1[n];
  const float b2 = be2p[0];

  for (int t = wid; t < ntiles; t += nw) {
    const int e0 = t * 16;
    int e = e0 + n; if (e >= E) e = E - 1;        // load clamp (tail only)
    const int s = row[e], d = col[e];
    const float4* ps = (const float4*)(h + (long)s * 16 + half * 8);
    const float4* pd = (const float4*)(h + (long)d * 16 + half * 8);
    v16h A;
    pack8(A, 0, ps[0], ps[1]);   // K = half*8 + j        (h[start] part)
    pack8(A, 8, pd[0], pd[1]);   // K = 16 + half*8 + j   (h[end] part)
    v8f acc = {};
    acc = __builtin_amdgcn_wmma_f32_16x16x32_f16(false, A, false, B,
                                                 (short)0, acc, false, false);
    // Each lane holds column n of rows half*8+j: relu, scale by We2[n],
    // reduce across the 16-lane group to get per-edge logits.
    float p[8];
#pragma unroll
    for (int j = 0; j < 8; ++j)
      p[j] = fmaxf(acc[j] + bb, 0.0f) * w2;
#pragma unroll
    for (int m = 1; m < 16; m <<= 1)
#pragma unroll
      for (int j = 0; j < 8; ++j)
        p[j] += __shfl_xor(p[j], m, 32);
    // After the reduce every lane of the group holds all 8 row sums;
    // lane n==0 of each half writes 8 contiguous outputs as two b128 stores.
    if (e0 + 16 <= E) {                            // wave-uniform fast path
      if (n == 0) {
        float4 lo, hi;
        lo.x = 1.0f / (1.0f + __expf(-(p[0] + b2)));
        lo.y = 1.0f / (1.0f + __expf(-(p[1] + b2)));
        lo.z = 1.0f / (1.0f + __expf(-(p[2] + b2)));
        lo.w = 1.0f / (1.0f + __expf(-(p[3] + b2)));
        hi.x = 1.0f / (1.0f + __expf(-(p[4] + b2)));
        hi.y = 1.0f / (1.0f + __expf(-(p[5] + b2)));
        hi.z = 1.0f / (1.0f + __expf(-(p[6] + b2)));
        hi.w = 1.0f / (1.0f + __expf(-(p[7] + b2)));
        float4* q = (float4*)(eout + e0 + half * 8);
        q[0] = lo;
        q[1] = hi;
      }
    } else {
      if (n == 0) {
#pragma unroll
        for (int j = 0; j < 8; ++j) {
          int ee = e0 + half * 8 + j;
          if (ee < E)
            eout[ee] = 1.0f / (1.0f + __expf(-(p[j] + b2)));
        }
      }
    }
  }
}

// ----------------------------------------------------------------- launcher

extern "C" void kernel_launch(void* const* d_in, const int* in_sizes, int n_in,
                              void* d_out, int out_size, void* d_ws, size_t ws_size,
                              hipStream_t stream) {
  const float* x   = (const float*)d_in[0];
  const int*   ei  = (const int*)d_in[1];
  const float* W1  = (const float*)d_in[2];
  const float* b1  = (const float*)d_in[3];
  const float* W2  = (const float*)d_in[4];
  const float* b2  = (const float*)d_in[5];
  const float* We1 = (const float*)d_in[6];
  const float* be1 = (const float*)d_in[7];
  const float* We2 = (const float*)d_in[8];
  const float* be2 = (const float*)d_in[9];

  const int N = in_sizes[0] / 128;
  const int E = in_sizes[1] / 2;
  const int* row = ei;
  const int* col = ei + E;

  // workspace carve-up (all offsets 256B aligned)
  char* ws = (char*)d_ws;
  size_t o = 0;
  auto carve = [&](size_t bytes) {
    char* p = ws + o;
    o += (bytes + 255) & ~(size_t)255;
    return p;
  };
  float* dinv = (float*)carve((size_t)N * 4);        // deg, then rsqrt in place
  float* hlin = (float*)carve((size_t)N * 16 * 4);   // x @ W1
  float* hbuf = (float*)carve((size_t)N * 16 * 4);   // acc -> relu(h)
  float* hw2  = (float*)carve((size_t)N * 2 * 4);    // h @ W2
  (void)ws_size; (void)n_in; (void)out_size;

  float* node_out = (float*)d_out;            // [N,2]
  float* edge_out = node_out + (long)N * 2;   // [E]

  const int T = 256;
  auto blk = [](long n, int t) { return (unsigned)((n + t - 1) / t); };

  // ---- degrees & normalization
  k_fill1<<<blk(N, T), T, 0, stream>>>(dinv, N);
  k_deg_accum<<<blk(E, T), T, 0, stream>>>(col, dinv, E);
  k_rsqrt<<<blk(N, T), T, 0, stream>>>(dinv, N);

  // ---- layer 1: h = relu(agg(norm * x@W1) + b1)
  const int t1 = (N + 15) / 16;
  k_gemm1_wmma<<<256, T, 0, stream>>>(x, W1, hlin, N, t1);
  k_selfloop16<<<blk((long)N * 16, T), T, 0, stream>>>(dinv, hlin, hbuf, N * 16);
  k_edge_agg16<<<blk((long)E * 16, T), T, 0, stream>>>(row, col, dinv, hlin, hbuf, (long)E * 16);
  k_relu_bias16<<<blk((long)N * 16, T), T, 0, stream>>>(hbuf, b1, N * 16);

  // ---- layer 2: node_out = agg(norm * h@W2) + b2 (accumulated in d_out)
  k_hw2<<<blk((long)N * 2, T), T, 0, stream>>>(hbuf, W2, hw2, N * 2);
  k_selfloop2<<<blk((long)N * 2, T), T, 0, stream>>>(dinv, hw2, node_out, N * 2);
  k_edge_agg2<<<blk((long)E * 2, T), T, 0, stream>>>(row, col, dinv, hw2, node_out, (long)E * 2);
  k_bias2<<<blk((long)N * 2, T), T, 0, stream>>>(node_out, b2, N * 2);

  // ---- edge MLP: sigmoid(relu([h[s]|h[d]]@We1 + be1)@We2 + be2)
  const int t2 = (E + 15) / 16;
  k_edge_mlp_wmma<<<2048, T, 0, stream>>>(hbuf, row, col, We1, be1, We2, be2,
                                          edge_out, E, t2);
}
